// SMED_85495618995018
// MI455X (gfx1250) — matrix operand order repaired and checked by reading, not compiled, not run on hardware
//
#include <hip/hip_runtime.h>
#include <math.h>

// ---------------------------------------------------------------------------
// MI455X (gfx1250): all large GEMMs via v_wmma_f32_16x16x32_bf16.
// All GEMM A-operands and weights live in bf16 (activations written bf16 by
// their producers). The FULLK GEMM instantiation (K % 32 == 0: every large
// GEMM here) has an unguarded inner loop:
//   per 32-K step / wave: 2x global_load_b128 (A) + 8x ds_load_b128 (B LDS)
//   + 4x v_wmma, no predication, no conversions.
// ---------------------------------------------------------------------------

typedef __attribute__((ext_vector_type(16))) __bf16 v16bf;
typedef __attribute__((ext_vector_type(8)))  float  v8f;
typedef __bf16 bf16;

#define Bsz  32
#define Tsz  1000
#define NNsz 512
#define NBsz 16
#define CTXs 64
#define EMBs 256
#define DCAT 768
#define Hsz  256
#define Lsz  4
#define PNsz 512
#define PBsz 2

__device__ __forceinline__ bf16 f2bf(float f) { return (bf16)f; }

__device__ __forceinline__ float bf2f(bf16 b) {
  union { bf16 b; unsigned short s; } in; in.b = b;
  union { unsigned u; float f; } out; out.u = ((unsigned)in.s) << 16;
  return out.f;
}

// ---------------------------------------------------------------------------
// Batched GEMM:  C[z][m,n] = act( sum_k A[z][m,k]*W[sel(z)][n,k]
//                                 + bias[sel(z)][n] + dsk[n]*U[m,n] )
// A bf16 [M,lda], W bf16 row-major [N,K]. Block tile 128x64 (8 wave32),
// wave tile 16x64. Outputs: Cf (f32) and/or Cbf (bf16), both optional.
// act: 0=none, 1=relu, 2=softplus.  FULLK: K % 32 == 0 (no k-tail guards).
// ---------------------------------------------------------------------------
template <bool FULLK>
__global__ __launch_bounds__(256)
void gemm_bf16_wmma(const bf16* __restrict__ A, int lda, long long aStride,
                    const bf16* __restrict__ W, const int* __restrict__ sel,
                    const float* __restrict__ bias,
                    float* __restrict__ Cf, bf16* __restrict__ Cbf,
                    int ldc, long long cStride,
                    const bf16* __restrict__ U, const float* __restrict__ dsk,
                    int ldu, int M, int N, int K, int act) {
  __shared__ bf16 Bt[64 * 40];   // 64 N-rows x 32 K, padded stride 40 bf16

  const int lane = threadIdx.x & 31;
  const int wave = threadIdx.x >> 5;
  const int z    = blockIdx.z;
  const int n0b  = blockIdx.y * 64;
  const int m0   = blockIdx.x * 128 + wave * 16;

  const bf16* Ab = A + (long long)z * aStride;
  const int sidx = sel ? sel[z] : 0;
  const bf16*  Wsel = W + (size_t)sidx * (size_t)N * (size_t)K;
  const float* bSel = bias ? bias + (size_t)sidx * (size_t)N : nullptr;

  // A-matrix 16x32 bf16 lane layout: m = lane&15; K-halves split by lane>=16.
  int mrow = m0 + (lane & 15);
  if (mrow > M - 1) mrow = M - 1;                  // clamp (stores masked)
  const bf16* Arow = Ab + (size_t)mrow * (size_t)lda;
  const int klo = (lane >> 4) * 8;

  // weight staging coords: thread t -> (n = t/4, 8-elem chunk t&3)
  const int sn = threadIdx.x >> 2;
  const int sc = threadIdx.x & 3;
  const bf16* Wrow = Wsel + (size_t)(n0b + sn) * (size_t)K + sc * 8;

  v8f acc[4] = {};

  for (int kk = 0; kk < K; kk += 32) {
    __syncthreads();
    {   // stage 64x32 bf16 weight tile into LDS
      uint4 val;
      if (FULLK) {
        val = *(const uint4*)(Wrow + kk);
      } else {
        val = make_uint4(0u, 0u, 0u, 0u);
        if (kk + sc * 8 + 8 <= K) val = *(const uint4*)(Wrow + kk);
      }
      *(uint4*)(&Bt[sn * 40 + sc * 8]) = val;
    }
    __syncthreads();

    // A fragment: two 16B chunks, no conversion (A already bf16)
    union { v16bf v; uint4 u[2]; } af;
    if (FULLK) {
      af.u[0] = *(const uint4*)(Arow + kk + klo);
      af.u[1] = *(const uint4*)(Arow + kk + 16 + klo);
    } else {
      af.u[0] = (kk + klo + 8 <= K)
                    ? *(const uint4*)(Arow + kk + klo) : make_uint4(0u,0u,0u,0u);
      af.u[1] = (kk + 16 + klo + 8 <= K)
                    ? *(const uint4*)(Arow + kk + 16 + klo) : make_uint4(0u,0u,0u,0u);
    }

#pragma unroll
    for (int j = 0; j < 4; ++j) {
      // B-matrix 32x16 lane layout: n = lane&15, k = (lane>>4)*8 + {0..7, 16..23}
      union { v16bf v; uint4 u[2]; } bfr;
      const uint4* p =
          (const uint4*)(&Bt[(j * 16 + (lane & 15)) * 40 + (lane >> 4) * 8]);
      bfr.u[0] = p[0];
      bfr.u[1] = p[2];
      acc[j] = __builtin_amdgcn_wmma_f32_16x16x32_bf16(
          false, af.v, false, bfr.v, (short)0, acc[j], false, false);
    }
  }

  // C/D layout: VGPR r -> m = m0 + (lane>>4)*8 + r ; n = lane&15
#pragma unroll
  for (int j = 0; j < 4; ++j) {
#pragma unroll
    for (int r = 0; r < 8; ++r) {
      int m = m0 + (lane >> 4) * 8 + r;
      if (m < M) {
        int n = n0b + j * 16 + (lane & 15);
        float v = acc[j][r];
        if (bSel) v += bSel[n];
        if (dsk)  v += dsk[n] * bf2f(U[(size_t)m * (size_t)ldu + n]);
        if (act == 1)      v = fmaxf(v, 0.f);
        else if (act == 2) v = (v > 20.f) ? v : log1pf(expf(v));
        if (Cf)  Cf[(long long)z * cStride + (size_t)m * (size_t)ldc + n] = v;
        if (Cbf) Cbf[(long long)z * cStride + (size_t)m * (size_t)ldc + n] = f2bf(v);
      }
    }
  }
}

// dispatch helper: pick FULLK instantiation from K
static inline void launch_gemm(dim3 grid, dim3 blk, hipStream_t stream,
                               const bf16* A, int lda, long long aStride,
                               const bf16* W, const int* sel, const float* bias,
                               float* Cf, bf16* Cbf, int ldc, long long cStride,
                               const bf16* U, const float* dsk, int ldu,
                               int M, int N, int K, int act) {
  if ((K & 31) == 0)
    gemm_bf16_wmma<true><<<grid, blk, 0, stream>>>(A, lda, aStride, W, sel, bias,
        Cf, Cbf, ldc, cStride, U, dsk, ldu, M, N, K, act);
  else
    gemm_bf16_wmma<false><<<grid, blk, 0, stream>>>(A, lda, aStride, W, sel, bias,
        Cf, Cbf, ldc, cStride, U, dsk, ldu, M, N, K, act);
}

// --------------------------- prep / small kernels ---------------------------

__global__ void cvt_bf16_kernel(const float* __restrict__ in,
                                bf16* __restrict__ o, int n) {
  int i = blockIdx.x * blockDim.x + threadIdx.x;
  if (i < n) o[i] = f2bf(in[i]);
}

// Wbu[l][n,k]: n<256 -> B_re[l,n,k]*exp(gamma[l,n]); n>=256 -> B_im*g
__global__ void build_wbu_kernel(const float* __restrict__ B_re,
                                 const float* __restrict__ B_im,
                                 const float* __restrict__ gamma_log,
                                 bf16* __restrict__ o) {
  int i = blockIdx.x * blockDim.x + threadIdx.x;
  const int per = 512 * DCAT;
  if (i >= Lsz * per) return;
  int l = i / per, r = i % per;
  int n = r / DCAT, k = r % DCAT;
  int h = (n < Hsz) ? n : n - Hsz;
  float g = expf(gamma_log[l * Hsz + h]);
  float v = (n < Hsz) ? B_re[((size_t)l * Hsz + h) * DCAT + k]
                      : B_im[((size_t)l * Hsz + h) * DCAT + k];
  o[i] = f2bf(v * g);
}

// Wy[l][o,k]: k<256 -> C_re[l,o,k]; k>=256 -> -C_im[l,o,k-256]
__global__ void build_wy_kernel(const float* __restrict__ C_re,
                                const float* __restrict__ C_im,
                                bf16* __restrict__ o) {
  int i = blockIdx.x * blockDim.x + threadIdx.x;
  const int per = DCAT * 512;
  if (i >= Lsz * per) return;
  int l = i / per, r = i % per;
  int oo = r / 512, hp = r % 512;
  float v = (hp < Hsz) ? C_re[((size_t)l * DCAT + oo) * Hsz + hp]
                       : -C_im[((size_t)l * DCAT + oo) * Hsz + (hp - Hsz)];
  o[i] = f2bf(v);
}

__global__ void ctx_kernel(const float* __restrict__ st,
                           const float* __restrict__ cw,
                           const float* __restrict__ cb,
                           const int* __restrict__ sid,
                           bf16* __restrict__ ctx) {
  int i = blockIdx.x * blockDim.x + threadIdx.x;
  if (i >= Bsz * EMBs) return;
  int b = i >> 8, e = i & 255;
  const float* row = st + (size_t)sid[b] * CTXs;
  float s = cb[e];
#pragma unroll 8
  for (int k = 0; k < CTXs; ++k) s += row[k] * cw[e * CTXs + k];
  ctx[i] = f2bf(fmaxf(s, 0.f));
}

__global__ void ctx_bcast_kernel(const bf16* __restrict__ ctx,
                                 bf16* __restrict__ h) {
  int i = blockIdx.x * blockDim.x + threadIdx.x;
  if (i >= Bsz * Tsz * EMBs) return;
  int b = i / (Tsz * EMBs);
  int r = i % (Tsz * EMBs);
  int t = r / EMBs, e = r % EMBs;
  h[((size_t)(b * Tsz + t)) * DCAT + 2 * EMBs + e] = ctx[b * EMBs + e];
}

// LRU diagonal complex scan: x_t = lam*x_{t-1} + Bu_t over 8192 chains.
// Reads f32 Bu, writes bf16 xs (GEMM operand format).
__global__ void lru_scan_kernel(const float* __restrict__ nu_log,
                                const float* __restrict__ theta_log,
                                const float* __restrict__ Bu,
                                bf16* __restrict__ xs, int layer) {
  int i = blockIdx.x * blockDim.x + threadIdx.x;
  if (i >= Bsz * Hsz) return;
  int b = i >> 8, h = i & 255;
  float nu  = expf(nu_log[layer * Hsz + h]);
  float th  = expf(theta_log[layer * Hsz + h]);
  float mag = expf(-nu);
  float lr = mag * cosf(th);
  float li = mag * sinf(th);
  float xr = 0.f, xi = 0.f;
  size_t base = (size_t)b * Tsz * 512 + h;
  for (int t = 0; t < Tsz; ++t) {
    size_t idx = base + (size_t)t * 512;
    float br = Bu[idx], bi = Bu[idx + Hsz];
    float nr = fmaf(lr, xr, fmaf(-li, xi, br));
    float ni = fmaf(lr, xi, fmaf(li, xr, bi));
    xr = nr; xi = ni;
    xs[idx] = f2bf(xr); xs[idx + Hsz] = f2bf(xi);
  }
}

// pred_b: tiny N=2 decoder, scalar dot products over bf16 h
__global__ void pred_b_kernel(const bf16* __restrict__ h,
                              const float* __restrict__ W,
                              const float* __restrict__ bias,
                              const int* __restrict__ sub,
                              float* __restrict__ out) {
  int i = blockIdx.x * blockDim.x + threadIdx.x;
  if (i >= Bsz * Tsz * PBsz) return;
  int m = i >> 1, o = i & 1;
  int b = m / Tsz;
  int s = sub[b];
  float acc = bias[s * PBsz + o];
  const bf16*  hr = h + (size_t)m * DCAT;
  const float* wr = W + ((size_t)s * PBsz + o) * DCAT;
#pragma unroll 8
  for (int k = 0; k < DCAT; ++k) acc = fmaf(bf2f(hr[k]), wr[k], acc);
  out[i] = acc;
}

// ---------------------------------------------------------------------------

extern "C" void kernel_launch(void* const* d_in, const int* in_sizes, int n_in,
                              void* d_out, int out_size, void* d_ws,
                              size_t ws_size, hipStream_t stream) {
  const float* neural_input   = (const float*)d_in[0];
  const float* behavior_input = (const float*)d_in[1];
  const int*   session_id     = (const int*)d_in[2];
  const int*   subject_id     = (const int*)d_in[3];
  const float* session_table  = (const float*)d_in[4];
  const float* neural_W       = (const float*)d_in[5];
  const float* neural_b       = (const float*)d_in[6];
  const float* behavior_W     = (const float*)d_in[7];
  const float* behavior_b     = (const float*)d_in[8];
  const float* ctx_W          = (const float*)d_in[9];
  const float* ctx_b          = (const float*)d_in[10];
  const float* nu_log         = (const float*)d_in[11];
  const float* theta_log      = (const float*)d_in[12];
  const float* gamma_log      = (const float*)d_in[13];
  const float* B_re           = (const float*)d_in[14];
  const float* B_im           = (const float*)d_in[15];
  const float* C_re           = (const float*)d_in[16];
  const float* C_im           = (const float*)d_in[17];
  const float* D_skip         = (const float*)d_in[18];
  const float* dec_n_W        = (const float*)d_in[19];
  const float* dec_n_b        = (const float*)d_in[20];
  const float* dec_b_W        = (const float*)d_in[21];
  const float* dec_b_b        = (const float*)d_in[22];
  float* out = (float*)d_out;
  (void)in_sizes; (void)n_in; (void)out_size; (void)ws_size;

  // workspace layout
  char* ws = (char*)d_ws;
  size_t off = 0;
  auto wsalloc = [&](size_t bytes) -> char* {
    char* p = ws + off;
    off += (bytes + 255) & ~(size_t)255;
    return p;
  };
  const size_t MT = (size_t)Bsz * Tsz;                 // 32000
  float* Bu   = (float*)wsalloc(MT * 512 * 4);         // scan input (f32)
  bf16* xsb   = (bf16*)wsalloc(MT * 512 * 2);          // scan output (bf16)
  bf16* hbfA  = (bf16*)wsalloc(MT * DCAT * 2);         // activations, bf16 only
  bf16* hbfB  = (bf16*)wsalloc(MT * DCAT * 2);
  bf16* ninb  = (bf16*)wsalloc(MT * NNsz * 2);         // bf16 neural_input
  bf16* binb  = (bf16*)wsalloc(MT * NBsz * 2);         // bf16 behavior_input
  bf16* ctx   = (bf16*)wsalloc((size_t)Bsz * EMBs * 2);
  bf16* nWb   = (bf16*)wsalloc((size_t)8 * EMBs * NNsz * 2);
  bf16* bWb   = (bf16*)wsalloc((size_t)8 * EMBs * NBsz * 2);
  bf16* dnWb  = (bf16*)wsalloc((size_t)8 * PNsz * DCAT * 2);
  bf16* wbu   = (bf16*)wsalloc((size_t)Lsz * 512 * DCAT * 2);
  bf16* wy    = (bf16*)wsalloc((size_t)Lsz * DCAT * 512 * 2);

  dim3 blk(256);
  auto g1 = [](long long n) { return dim3((unsigned)((n + 255) / 256)); };

  // ---- prep: bf16-pack weights and network inputs (once per call) ----
  cvt_bf16_kernel<<<g1(MT * NNsz), blk, 0, stream>>>(neural_input, ninb, (int)(MT * NNsz));
  cvt_bf16_kernel<<<g1(MT * NBsz), blk, 0, stream>>>(behavior_input, binb, (int)(MT * NBsz));
  cvt_bf16_kernel<<<g1(8 * EMBs * NNsz), blk, 0, stream>>>(neural_W, nWb, 8 * EMBs * NNsz);
  cvt_bf16_kernel<<<g1(8 * EMBs * NBsz), blk, 0, stream>>>(behavior_W, bWb, 8 * EMBs * NBsz);
  cvt_bf16_kernel<<<g1(8 * PNsz * DCAT), blk, 0, stream>>>(dec_n_W, dnWb, 8 * PNsz * DCAT);
  build_wbu_kernel<<<g1(Lsz * 512 * DCAT), blk, 0, stream>>>(B_re, B_im, gamma_log, wbu);
  build_wy_kernel<<<g1(Lsz * DCAT * 512), blk, 0, stream>>>(C_re, C_im, wy);
  ctx_kernel<<<g1(Bsz * EMBs), blk, 0, stream>>>(session_table, ctx_W, ctx_b, session_id, ctx);

  // ---- embedders: h[:, 0:256]=en, [256:512]=eb, [512:768]=ctx (bf16) ----
  launch_gemm(dim3(8, 4, Bsz), blk, stream,
      ninb, NNsz, (long long)Tsz * NNsz, nWb, subject_id, neural_b,
      nullptr, hbfA, DCAT, (long long)Tsz * DCAT, nullptr, nullptr, 0,
      Tsz, EMBs, NNsz, /*relu*/ 1);
  launch_gemm(dim3(8, 4, Bsz), blk, stream,
      binb, NBsz, (long long)Tsz * NBsz, bWb, subject_id, behavior_b,
      nullptr, hbfA + EMBs, DCAT, (long long)Tsz * DCAT, nullptr, nullptr, 0,
      Tsz, EMBs, NBsz, /*relu*/ 1);
  ctx_bcast_kernel<<<g1(MT * EMBs), blk, 0, stream>>>(ctx, hbfA);

  // ---- 4 LRU layers ----
  bf16* hcur = hbfA;
  bf16* hnxt = hbfB;
  for (int l = 0; l < Lsz; ++l) {
    // Bu = h @ [B_re*g ; B_im*g]^T  -> [32000, 512] f32 (re | im)
    launch_gemm(dim3(250, 8, 1), blk, stream,
        hcur, DCAT, 0, wbu + (size_t)l * 512 * DCAT, nullptr, nullptr,
        Bu, nullptr, 512, 0, nullptr, nullptr, 0, (int)MT, 512, DCAT, 0);
    // sequential diagonal recurrence (f32 in, bf16 out)
    lru_scan_kernel<<<g1(Bsz * Hsz), blk, 0, stream>>>(nu_log, theta_log, Bu, xsb, l);
    // h' = Re(xs @ C) + D_skip * h  -> [32000, 768] bf16
    launch_gemm(dim3(250, 12, 1), blk, stream,
        xsb, 512, 0, wy + (size_t)l * DCAT * 512, nullptr, nullptr,
        nullptr, hnxt, DCAT, 0, hcur, D_skip + (size_t)l * DCAT, DCAT,
        (int)MT, DCAT, 512, 0);
    bf16* t = hcur; hcur = hnxt; hnxt = t;
  }

  // ---- decoders ----
  launch_gemm(dim3(8, 8, Bsz), blk, stream,
      hcur, DCAT, (long long)Tsz * DCAT, dnWb, subject_id, dec_n_b,
      out, nullptr, PNsz, (long long)Tsz * PNsz, nullptr, nullptr, 0,
      Tsz, PNsz, DCAT, /*softplus*/ 2);
  pred_b_kernel<<<g1(MT * PBsz), blk, 0, stream>>>(
      hcur, dec_b_W, dec_b_b, subject_id, out + MT * PNsz);
}
